// HTGAT_15418932593211
// MI455X (gfx1250) — matrix operand order repaired and checked by reading, not compiled, not run on hardware
//
#include <hip/hip_runtime.h>
#include <hip/hip_bf16.h>

typedef __bf16 bf16;
typedef __attribute__((ext_vector_type(16))) __bf16 v16bf;
typedef __attribute__((ext_vector_type(8)))  float   v8f;

#define NN 50000      // nodes
#define NE 400000     // edges
#define HC 256        // heads*dim (both layers)
#define NT 3          // node types
#define TEt 4         // edge types
#define NH 4          // heads
#define EDIM 32       // edge feature dim

union FragBF { v16bf v; uint4 u[2]; };

__device__ __forceinline__ unsigned encF(float f) {
  unsigned u = __float_as_uint(f);
  return (u & 0x80000000u) ? ~u : (u | 0x80000000u);   // monotonic float->uint
}
__device__ __forceinline__ float decF(unsigned u) {
  unsigned v = (u & 0x80000000u) ? (u & 0x7FFFFFFFu) : ~u;
  return __uint_as_float(v);
}

// ---------- prep kernels ----------
__global__ void k_cvt_bf16(const float* __restrict__ in, bf16* __restrict__ out, int n) {
  int i = blockIdx.x * blockDim.x + threadIdx.x;
  if (i < n) out[i] = (bf16)in[i];
}

// W (t,k,n) f32 -> Wt (t,n,k) bf16   (so B-fragment loads are contiguous in K)
__global__ void k_w_transpose(const float* __restrict__ W, bf16* __restrict__ Wt) {
  int i = blockIdx.x * blockDim.x + threadIdx.x;       // NT*256*256
  if (i >= NT * HC * HC) return;
  int t = i >> 16;
  int r = i & 65535;
  int n = r >> 8;
  int k = r & 255;
  Wt[i] = (bf16)W[t * 65536 + k * HC + n];
}

__global__ void k_zero(float* __restrict__ p, int n) {
  int i = blockIdx.x * blockDim.x + threadIdx.x;
  if (i < n) p[i] = 0.f;
}

// ---------- type-conditioned GEMM: Out[row] = A[row] @ W[type[row]] ----------
// One wave per 16x16 output tile, K=256 in 8 steps of 32, bf16 WMMA,
// 3 accumulators (one per node type), per-row select in epilogue.
__global__ __launch_bounds__(256)
void k_gemm_typed(const bf16* __restrict__ A, const bf16* __restrict__ Wt,
                  const int* __restrict__ ntype, float* __restrict__ Out) {
  int tile = blockIdx.x * 8 + (threadIdx.x >> 5);      // 50000 tiles total
  int lane = threadIdx.x & 31;
  int m0  = (tile >> 4) << 4;
  int n0  = (tile & 15) << 4;
  int l16 = lane & 15;
  int hi  = lane >> 4;

  v8f acc0 = {}, acc1 = {}, acc2 = {};
  const bf16* arow = A + (size_t)(m0 + l16) * HC;
  const bf16* b0p  = Wt + (size_t)(n0 + l16) * HC;     // type 0
  const bf16* b1p  = b0p + HC * HC;                    // type 1
  const bf16* b2p  = b1p + HC * HC;                    // type 2

#pragma unroll
  for (int k0 = 0; k0 < HC; k0 += 32) {
    FragBF fa, f0, f1, f2;
    // A 16x32 bf16, contiguous k order used here (same as B): lanes0-15 K=k0..k0+15, lanes16-31 K=k0+16..k0+31
    fa.u[0] = *(const uint4*)(arow + k0 + hi * 16);
    fa.u[1] = *(const uint4*)(arow + k0 + hi * 16 + 8);
    // B 32x16 bf16, contiguous k order used here (same as A): lane = column; lanes0-15 K=k0..k0+15, lanes16-31 K=k0+16..k0+31
    f0.u[0] = *(const uint4*)(b0p + k0 + hi * 16);
    f0.u[1] = *(const uint4*)(b0p + k0 + hi * 16 + 8);
    f1.u[0] = *(const uint4*)(b1p + k0 + hi * 16);
    f1.u[1] = *(const uint4*)(b1p + k0 + hi * 16 + 8);
    f2.u[0] = *(const uint4*)(b2p + k0 + hi * 16);
    f2.u[1] = *(const uint4*)(b2p + k0 + hi * 16 + 8);
    acc0 = __builtin_amdgcn_wmma_f32_16x16x32_bf16(false, fa.v, false, f0.v, (short)0, acc0, false, false);
    acc1 = __builtin_amdgcn_wmma_f32_16x16x32_bf16(false, fa.v, false, f1.v, (short)0, acc1, false, false);
    acc2 = __builtin_amdgcn_wmma_f32_16x16x32_bf16(false, fa.v, false, f2.v, (short)0, acc2, false, false);
  }

#pragma unroll
  for (int r = 0; r < 8; ++r) {
    int row = m0 + r + hi * 8;                         // C layout: VGPR r -> M=r / r+8
    int t = ntype[row];
    float v = (t == 0) ? acc0[r] : (t == 1) ? acc1[r] : acc2[r];
    Out[(size_t)row * HC + n0 + l16] = v;
  }
}

// ---------- attention scores s[n,t,h] = sum_c P[n,h*64+c]*att[t,h,c] ----------
__global__ void k_att_score(const float* __restrict__ P, const float* __restrict__ att,
                            float* __restrict__ S) {
  int i = blockIdx.x * blockDim.x + threadIdx.x;
  if (i >= NN * 16) return;
  int n = i >> 4;
  int t = (i >> 2) & 3;
  int h = i & 3;
  const float* pr = P + (size_t)n * HC + h * 64;
  const float* ar = att + (t * NH + h) * 64;
  float acc = 0.f;
#pragma unroll 8
  for (int c = 0; c < 64; ++c) acc += pr[c] * ar[c];
  S[i] = acc;
}

// v[t,d,h] = sum_c W_edge[t,d,h*64+c] * att_edge[t,h,c]   (512 values)
__global__ void k_vedge(const float* __restrict__ We, const float* __restrict__ attE,
                        float* __restrict__ V) {
  int i = blockIdx.x * blockDim.x + threadIdx.x;       // TEt*EDIM*NH = 512
  if (i >= TEt * EDIM * NH) return;
  int t = i >> 7;
  int d = (i >> 2) & 31;
  int h = i & 3;
  const float* wr = We + ((size_t)(t * EDIM + d)) * HC + h * 64;
  const float* ar = attE + (t * NH + h) * 64;
  float acc = 0.f;
#pragma unroll 8
  for (int c = 0; c < 64; ++c) acc += wr[c] * ar[c];
  V[i] = acc;
}

// alpha + leaky relu + segment-max (encoded atomicMax)
__global__ void k_edge_alpha(const int* __restrict__ ei, const int* __restrict__ etype,
                             const float* __restrict__ eattr,
                             const float* __restrict__ Ssrc, const float* __restrict__ Sdst,
                             const float* __restrict__ V,
                             float* __restrict__ Alpha, unsigned* __restrict__ AmaxEnc) {
  int i = blockIdx.x * blockDim.x + threadIdx.x;
  if (i >= NE * 4) return;
  int e = i >> 2, h = i & 3;
  int s = ei[e], d = ei[NE + e], t = etype[e];
  float a = Ssrc[s * 16 + t * 4 + h] + Sdst[d * 16 + t * 4 + h];
  const float* ea = eattr + (size_t)e * EDIM;
  const float* vp = V + t * 128 + h;                   // stride 4 over d
  float acc = 0.f;
#pragma unroll
  for (int dd = 0; dd < EDIM; ++dd) acc += ea[dd] * vp[dd * 4];
  a += acc;
  a = (a >= 0.f) ? a : 0.2f * a;
  Alpha[i] = a;
  atomicMax(AmaxEnc + d * 4 + h, encF(a));
}

// decode amax in-place (enc==0 => no edges => 0, matches isfinite->0), init denom
__global__ void k_amax_denom(unsigned* __restrict__ A, float* __restrict__ Denom, int n) {
  int i = blockIdx.x * blockDim.x + threadIdx.x;
  if (i >= n) return;
  unsigned u = A[i];
  float m = (u == 0u) ? 0.f : decF(u);
  if (!(m >= -3.0e38f && m <= 3.0e38f)) m = 0.f;
  ((float*)A)[i] = m;
  Denom[i] = 0.f;
}

// ae = exp(alpha - amax[dst]); denom += ae
__global__ void k_edge_exp(const int* __restrict__ ei, float* __restrict__ Alpha,
                           const float* __restrict__ Amax, float* __restrict__ Denom) {
  int i = blockIdx.x * blockDim.x + threadIdx.x;
  if (i >= NE * 4) return;
  int e = i >> 2, h = i & 3;
  int d = ei[NE + e];
  float ae = __expf(Alpha[i] - Amax[d * 4 + h]);
  Alpha[i] = ae;
  atomicAdd(Denom + d * 4 + h, ae);
}

// scatter aggregation: 64 threads/edge, 4 channels each (float4 load, 4 atomics)
__global__ void k_aggregate(const int* __restrict__ ei, const float* __restrict__ Psrc,
                            const float* __restrict__ Alpha, const float* __restrict__ Denom,
                            float* __restrict__ Agg) {
  int i = blockIdx.x * blockDim.x + threadIdx.x;       // NE*64
  if (i >= NE * 64) return;
  int e = i >> 6;
  int j = i & 63;                                      // channels 4j..4j+3
  int s = ei[e], d = ei[NE + e];
  int h = j >> 4;
  float attn = Alpha[e * 4 + h] / (Denom[d * 4 + h] + 1e-16f);
  float4 xv = *(const float4*)(Psrc + (size_t)s * HC + j * 4);
  float* op = Agg + (size_t)d * HC + j * 4;
  atomicAdd(op + 0, xv.x * attn);
  atomicAdd(op + 1, xv.y * attn);
  atomicAdd(op + 2, xv.z * attn);
  atomicAdd(op + 3, xv.w * attn);
}

// layer1 epilogue: +bias, LayerNorm, ELU, write bf16 input for layer 2
__global__ __launch_bounds__(256)
void k_epilogue1(const float* __restrict__ Agg, const float* __restrict__ bias,
                 const float* __restrict__ gamma, const float* __restrict__ beta,
                 bf16* __restrict__ Hbf) {
  __shared__ float sm[256];
  int n = blockIdx.x, c = threadIdx.x;
  float v = Agg[(size_t)n * HC + c] + bias[c];
  sm[c] = v; __syncthreads();
  for (int s = 128; s > 0; s >>= 1) { if (c < s) sm[c] += sm[c + s]; __syncthreads(); }
  float mu = sm[0] * (1.0f / HC);
  __syncthreads();
  float dv = v - mu;
  sm[c] = dv * dv; __syncthreads();
  for (int s = 128; s > 0; s >>= 1) { if (c < s) sm[c] += sm[c + s]; __syncthreads(); }
  float var = sm[0] * (1.0f / HC);
  float y = dv * rsqrtf(var + 1e-5f) * gamma[c] + beta[c];
  float e = (y > 0.f) ? y : (__expf(y) - 1.0f);        // ELU
  Hbf[(size_t)n * HC + c] = (bf16)e;
}

// layer2 epilogue: mean over heads + bias2
__global__ void k_epilogue2(const float* __restrict__ Agg, const float* __restrict__ bias,
                            float* __restrict__ out) {
  int i = blockIdx.x * blockDim.x + threadIdx.x;
  if (i >= NN * 64) return;
  int n = i >> 6, o = i & 63;
  const float* row = Agg + (size_t)n * HC + o;
  out[i] = 0.25f * (row[0] + row[64] + row[128] + row[192]) + bias[o];
}

// ---------- host ----------
static void run_layer(hipStream_t stream,
                      const bf16* bfA, const bf16* WtS, const bf16* WtD,
                      const int* ntype, const int* ei, const int* etype,
                      const float* eattr, const float* attS, const float* attD,
                      const float* Wedge, const float* attE,
                      float* Psrc, float* Pdst, float* Ssrc, float* Sdst,
                      float* V, float* Alpha, float* Amax, float* Denom, float* Agg) {
  k_gemm_typed<<<NN / 8, 256, 0, stream>>>(bfA, WtS, ntype, Psrc);   // 50000 tiles / 8 waves
  k_gemm_typed<<<NN / 8, 256, 0, stream>>>(bfA, WtD, ntype, Pdst);
  k_att_score<<<(NN * 16 + 255) / 256, 256, 0, stream>>>(Psrc, attS, Ssrc);
  k_att_score<<<(NN * 16 + 255) / 256, 256, 0, stream>>>(Pdst, attD, Sdst);
  k_vedge<<<2, 256, 0, stream>>>(Wedge, attE, V);
  k_zero<<<(NN * 4 + 255) / 256, 256, 0, stream>>>(Amax, NN * 4);    // enc sentinel 0
  k_edge_alpha<<<(NE * 4 + 255) / 256, 256, 0, stream>>>(ei, etype, eattr, Ssrc, Sdst, V,
                                                         Alpha, (unsigned*)Amax);
  k_amax_denom<<<(NN * 4 + 255) / 256, 256, 0, stream>>>((unsigned*)Amax, Denom, NN * 4);
  k_zero<<<(NN * HC + 255) / 256, 256, 0, stream>>>(Agg, NN * HC);
  k_edge_exp<<<(NE * 4 + 255) / 256, 256, 0, stream>>>(ei, Alpha, Amax, Denom);
  k_aggregate<<<(NE * 64 + 255) / 256, 256, 0, stream>>>(ei, Psrc, Alpha, Denom, Agg);
}

extern "C" void kernel_launch(void* const* d_in, const int* in_sizes, int n_in,
                              void* d_out, int out_size, void* d_ws, size_t ws_size,
                              hipStream_t stream) {
  const float* x       = (const float*)d_in[0];
  const int*   ei      = (const int*)d_in[1];
  const int*   ntype   = (const int*)d_in[2];
  const int*   etype   = (const int*)d_in[3];
  const float* eattr   = (const float*)d_in[4];
  const float* W_src1  = (const float*)d_in[5];
  const float* W_dst1  = (const float*)d_in[6];
  const float* attS1   = (const float*)d_in[7];
  const float* attD1   = (const float*)d_in[8];
  const float* Wedge1  = (const float*)d_in[9];
  const float* attE1   = (const float*)d_in[10];
  const float* bias1   = (const float*)d_in[11];
  const float* gamma1  = (const float*)d_in[12];
  const float* beta1   = (const float*)d_in[13];
  const float* W_src2  = (const float*)d_in[14];
  const float* W_dst2  = (const float*)d_in[15];
  const float* attS2   = (const float*)d_in[16];
  const float* attD2   = (const float*)d_in[17];
  const float* Wedge2  = (const float*)d_in[18];
  const float* attE2   = (const float*)d_in[19];
  const float* bias2   = (const float*)d_in[20];

  char* p = (char*)d_ws;
  auto take = [&](size_t bytes) -> char* {
    char* r = p; p += (bytes + 255) & ~(size_t)255; return r;
  };
  bf16*  bfA   = (bf16*)take((size_t)NN * HC * 2);          // x_bf16, reused as h_bf16
  bf16*  WtS1  = (bf16*)take((size_t)NT * HC * HC * 2);
  bf16*  WtD1  = (bf16*)take((size_t)NT * HC * HC * 2);
  bf16*  WtS2  = (bf16*)take((size_t)NT * HC * HC * 2);
  bf16*  WtD2  = (bf16*)take((size_t)NT * HC * HC * 2);
  float* Psrc  = (float*)take((size_t)NN * HC * 4);
  float* Pdst  = (float*)take((size_t)NN * HC * 4);
  float* Ssrc  = (float*)take((size_t)NN * 16 * 4);
  float* Sdst  = (float*)take((size_t)NN * 16 * 4);
  float* V     = (float*)take(512 * 4);
  float* Alpha = (float*)take((size_t)NE * 4 * 4);
  float* Amax  = (float*)take((size_t)NN * 4 * 4);
  float* Denom = (float*)take((size_t)NN * 4 * 4);
  float* Agg   = (float*)take((size_t)NN * HC * 4);

  const int wn = NT * HC * HC;
  k_cvt_bf16<<<(NN * HC + 255) / 256, 256, 0, stream>>>(x, bfA, NN * HC);
  k_w_transpose<<<(wn + 255) / 256, 256, 0, stream>>>(W_src1, WtS1);
  k_w_transpose<<<(wn + 255) / 256, 256, 0, stream>>>(W_dst1, WtD1);
  k_w_transpose<<<(wn + 255) / 256, 256, 0, stream>>>(W_src2, WtS2);
  k_w_transpose<<<(wn + 255) / 256, 256, 0, stream>>>(W_dst2, WtD2);

  run_layer(stream, bfA, WtS1, WtD1, ntype, ei, etype, eattr, attS1, attD1,
            Wedge1, attE1, Psrc, Pdst, Ssrc, Sdst, V, Alpha, Amax, Denom, Agg);
  k_epilogue1<<<NN, 256, 0, stream>>>(Agg, bias1, gamma1, beta1, bfA);
  run_layer(stream, bfA, WtS2, WtD2, ntype, ei, etype, eattr, attS2, attD2,
            Wedge2, attE2, Psrc, Pdst, Ssrc, Sdst, V, Alpha, Amax, Denom, Agg);
  k_epilogue2<<<(NN * 64 + 255) / 256, 256, 0, stream>>>(Agg, bias2, (float*)d_out);
}
